// GraphMemoryVQ_24902220382516
// MI455X (gfx1250) — matrix operand order repaired and checked by reading, not compiled, not run on hardware
//
#include <hip/hip_runtime.h>
#include <cstdint>

// Problem constants (from reference): B=16, S=2048, D=256, K=4096
#define M_TOTAL 32768   // B*S rows
#define KD      512     // 2*D dot dimension
#define NSYM    4096    // codebook entries
#define DLAT    256     // D

typedef __attribute__((ext_vector_type(16))) __bf16 v16bf;
typedef __attribute__((ext_vector_type(8)))  __bf16 v8bf;
typedef __attribute__((ext_vector_type(4)))  __bf16 v4bf;
typedef __attribute__((ext_vector_type(8)))  float  v8f;
typedef __attribute__((ext_vector_type(4)))  float  v4f;
typedef __attribute__((ext_vector_type(2)))  float  v2f;

// ---------------------------------------------------------------------------
// Async global->LDS copy of 64 contiguous bytes (4 x b128), ASYNCcnt-tracked.
// inst_offset applies to BOTH the global and LDS addresses (ISA 15.18.3).
// ---------------------------------------------------------------------------
__device__ __forceinline__ void async_copy64(uint32_t lds_addr, const void* gaddr) {
  asm volatile(
      "global_load_async_to_lds_b128 %0, %1, off\n\t"
      "global_load_async_to_lds_b128 %0, %1, off offset:16\n\t"
      "global_load_async_to_lds_b128 %0, %1, off offset:32\n\t"
      "global_load_async_to_lds_b128 %0, %1, off offset:48"
      :: "v"(lds_addr), "v"(gaddr) : "memory");
}
__device__ __forceinline__ void async_wait8()  { asm volatile("s_wait_asynccnt 0x8" ::: "memory"); }
__device__ __forceinline__ void async_wait0()  { asm volatile("s_wait_asynccnt 0x0" ::: "memory"); }

// ---------------------------------------------------------------------------
// Kernel 1a: split z_flat = [z_real | z_imag] into bf16 hi + residual lo
// ---------------------------------------------------------------------------
__global__ void convert_z_kernel(const float* __restrict__ z_real,
                                 const float* __restrict__ z_imag,
                                 __bf16* __restrict__ z_hi,
                                 __bf16* __restrict__ z_lo) {
  const int e = blockIdx.x * blockDim.x + threadIdx.x;   // 4-elem chunk id
  const int i = e * 4;
  const int m = i >> 9;           // row
  const int k = i & (KD - 1);     // col in [0,512), 4-chunk never straddles 256
  const float* src = (k < DLAT) ? (z_real + (size_t)m * DLAT + k)
                                : (z_imag + (size_t)m * DLAT + (k - DLAT));
  v4f f = *(const v4f*)src;
  v4bf hi, lo;
  #pragma unroll
  for (int t = 0; t < 4; ++t) {
    __bf16 h = (__bf16)f[t];
    hi[t] = h;
    lo[t] = (__bf16)(f[t] - (float)h);
  }
  *(v4bf*)(z_hi + (size_t)m * KD + k) = hi;
  *(v4bf*)(z_lo + (size_t)m * KD + k) = lo;
}

// ---------------------------------------------------------------------------
// Kernel 1b: split codebook into bf16 hi + residual lo
// ---------------------------------------------------------------------------
__global__ void convert_c_kernel(const float* __restrict__ codebook,
                                 __bf16* __restrict__ c_hi,
                                 __bf16* __restrict__ c_lo) {
  const int e = blockIdx.x * blockDim.x + threadIdx.x;
  const int i = e * 4;
  v4f f = *(const v4f*)(codebook + i);
  v4bf hi, lo;
  #pragma unroll
  for (int t = 0; t < 4; ++t) {
    __bf16 h = (__bf16)f[t];
    hi[t] = h;
    lo[t] = (__bf16)(f[t] - (float)h);
  }
  *(v4bf*)(c_hi + i) = hi;
  *(v4bf*)(c_lo + i) = lo;
}

// ---------------------------------------------------------------------------
// Kernel 1c: codebook row norms (fp32, exact)
// ---------------------------------------------------------------------------
__global__ void cnorm_kernel(const float* __restrict__ codebook,
                             float* __restrict__ cnorm) {
  int n = blockIdx.x * blockDim.x + threadIdx.x;
  if (n >= NSYM) return;
  const float* row = codebook + (size_t)n * KD;
  float s = 0.f;
  #pragma unroll 8
  for (int j = 0; j < KD; ++j) { float v = row[j]; s += v * v; }
  cnorm[n] = s;
}

// ---------------------------------------------------------------------------
// Kernel 2: fused distance-GEMM (bf16x3 split, WMMA) + graph bias + argmin.
//   WG = 256 threads = 8 waves; WG tile = 128 rows x all 4096 cols.
//   Wave tile = 16 rows x 128 cols per supertile -> 8 v8f accumulators.
//   Double-buffered async global->LDS staging, pipelined across (nsup, kc).
// ---------------------------------------------------------------------------
#define NSTAGE ((NSYM / 128) * (KD / 32))   // 32 * 16 = 512 stages
#define BUFBYTES 32768                       // 4 arrays * 128 rows * 64 B

__launch_bounds__(256)
__global__ void vq_argmin_kernel(const __bf16* __restrict__ z_hi,
                                 const __bf16* __restrict__ z_lo,
                                 const __bf16* __restrict__ c_hi,
                                 const __bf16* __restrict__ c_lo,
                                 const int*   __restrict__ prev_idx,
                                 const float* __restrict__ adjacency,
                                 const float* __restrict__ cnorm,
                                 int*         __restrict__ idx_out) {
  extern __shared__ __align__(16) char smem_raw[];   // 2 x 32 KB buffers
  const uint32_t lds_base = (uint32_t)(uintptr_t)(void*)smem_raw;

  const int tid  = threadIdx.x;
  const int lane = tid & 31;
  const int wid  = tid >> 5;      // wave owns rows wid*16 .. wid*16+15
  const int h    = lane >> 4;     // half-wave (WMMA K/M split)
  const int cl   = lane & 15;

  const int mblk = blockIdx.x * 128;

  // this thread's two copy slices: one A-array row, one B-array row
  const int sA_arr = tid >> 7;           // 0 = z_hi, 1 = z_lo
  const int sA_row = tid & 127;
  const int sB_arr = 2 + (tid >> 7);     // 2 = c_hi, 3 = c_lo
  const int sB_row = tid & 127;
  const __bf16* zsrc = (sA_arr == 0) ? z_hi : z_lo;
  const __bf16* csrc = (sB_arr == 2) ? c_hi : c_lo;

  // issue async copies for pipeline stage t into buffer `buf`
  auto issue_stage = [&](int t, int buf) {
    const int ns = t >> 4;               // supertile
    const int kc = t & 15;               // k-chunk
    const int nblk = ns * 128;
    const int k0   = kc * 32;
    const uint32_t lb = lds_base + (uint32_t)buf * BUFBYTES;
    async_copy64(lb + (uint32_t)sA_arr * 8192 + (uint32_t)sA_row * 64,
                 (const char*)zsrc + (size_t)(mblk + sA_row) * (KD * 2) + k0 * 2);
    async_copy64(lb + (uint32_t)sB_arr * 8192 + (uint32_t)sB_row * 64,
                 (const char*)csrc + (size_t)(nblk + sB_row) * (KD * 2) + k0 * 2);
  };

  // previous symbol for this lane's 8 rows (row m = mblk + wid*16 + 8h + v)
  int prevRow[8];
  #pragma unroll
  for (int v = 0; v < 8; ++v)
    prevRow[v] = prev_idx[mblk + wid * 16 + h * 8 + v];

  float minval[8];
  int   minidx[8];
  #pragma unroll
  for (int v = 0; v < 8; ++v) { minval[v] = 3.4e38f; minidx[v] = 0; }

  issue_stage(0, 0);
  int t = 0;

  for (int nsup = 0; nsup < NSYM / 128; ++nsup) {
    const int nblk = nsup * 128;

    v8f acc[8];
    #pragma unroll
    for (int ni = 0; ni < 8; ++ni) acc[ni] = v8f{};

    for (int kc = 0; kc < KD / 32; ++kc, ++t) {
      const int buf = t & 1;
      if (t + 1 < NSTAGE) { issue_stage(t + 1, 1 - buf); async_wait8(); }
      else                { async_wait0(); }
      __syncthreads();                       // stage t resident in LDS

      const __bf16* Ah = (const __bf16*)(smem_raw + buf * BUFBYTES);
      const __bf16* Al = Ah + 4096;
      const __bf16* Bh = Al + 4096;
      const __bf16* Bl = Bh + 4096;

      // A fragment: lane holds row (wid*16 + cl), 16 contiguous k at h*16
      const __bf16* ph = Ah + (wid * 16 + cl) * 32 + h * 16;
      const __bf16* pl = Al + (wid * 16 + cl) * 32 + h * 16;
      v8bf h0 = *(const v8bf*)ph;
      v8bf h1 = *(const v8bf*)(ph + 8);
      v8bf l0 = *(const v8bf*)pl;
      v8bf l1 = *(const v8bf*)(pl + 8);
      v16bf ah = __builtin_shufflevector(h0, h1, 0,1,2,3,4,5,6,7,8,9,10,11,12,13,14,15);
      v16bf al = __builtin_shufflevector(l0, l1, 0,1,2,3,4,5,6,7,8,9,10,11,12,13,14,15);

      #pragma unroll
      for (int ni = 0; ni < 8; ++ni) {
        // B fragment: lane holds col (ni*16 + cl), 16 contiguous k at h*16
        const int c = ni * 16 + cl;
        v16bf bh = *(const v16bf*)(Bh + c * 32 + h * 16);
        v16bf bl = *(const v16bf*)(Bl + c * 32 + h * 16);
        // bf16x3: hi*hi + hi*lo + lo*hi (lo*lo is below fp32 ulp here)
        acc[ni] = __builtin_amdgcn_wmma_f32_16x16x32_bf16(
            false, ah, false, bh, (short)0, acc[ni], false, false);
        acc[ni] = __builtin_amdgcn_wmma_f32_16x16x32_bf16(
            false, ah, false, bl, (short)0, acc[ni], false, false);
        acc[ni] = __builtin_amdgcn_wmma_f32_16x16x32_bf16(
            false, al, false, bh, (short)0, acc[ni], false, false);
      }
      __syncthreads();                       // all waves done reading buf
    }

    // epilogue: score = ||c||^2 - 2 z.c - 0.8*sigmoid(adj[prev, n]); argmin.
    // (||z||^2 is constant over n -> irrelevant to argmin, dropped)
    #pragma unroll
    for (int ni = 0; ni < 8; ++ni) {
      const int n  = nblk + ni * 16 + cl;
      const float cn = cnorm[n];
      #pragma unroll
      for (int v = 0; v < 8; ++v) {
        float g   = adjacency[(size_t)prevRow[v] * NSYM + n];
        float sig = 1.0f / (1.0f + __expf(-g));
        float sc2 = cn - 2.0f * acc[ni][v] - 0.8f * sig;
        if (sc2 < minval[v] || (sc2 == minval[v] && n < minidx[v])) {
          minval[v] = sc2; minidx[v] = n;
        }
      }
    }
  }

  // butterfly argmin across the 16 lanes of each half-wave (WMMA C-layout
  // keeps every output row inside one half, so offsets 1,2,4,8 suffice)
  #pragma unroll
  for (int off = 1; off < 16; off <<= 1) {
    #pragma unroll
    for (int v = 0; v < 8; ++v) {
      float ov = __shfl_xor(minval[v], off, 32);
      int   oi = __shfl_xor(minidx[v], off, 32);
      if (ov < minval[v] || (ov == minval[v] && oi < minidx[v])) {
        minval[v] = ov; minidx[v] = oi;
      }
    }
  }
  if (cl == 0) {
    #pragma unroll
    for (int v = 0; v < 8; ++v)
      idx_out[mblk + wid * 16 + h * 8 + v] = minidx[v];
  }
}

// ---------------------------------------------------------------------------
// Kernel 3: gather winning codeword -> interleaved complex output + loss
//   loss = loss_vq + 0.01*loss_commit = 1.01 * mean((z_q - z)^2);
//   z_q_ste == z_q numerically.  One wave per row.
// ---------------------------------------------------------------------------
__global__ void finalize_kernel(const float* __restrict__ z_real,
                                const float* __restrict__ z_imag,
                                const float* __restrict__ codebook,
                                const int*   __restrict__ idx_in,
                                float* __restrict__ out_z,     // [M][D] complex64
                                float* __restrict__ out_loss)  // [M]
{
  const int m    = blockIdx.x * (blockDim.x >> 5) + (threadIdx.x >> 5);
  const int lane = threadIdx.x & 31;
  const int idx  = idx_in[m];
  const float* c  = codebook + (size_t)idx * KD;
  const float* zr = z_real   + (size_t)m * DLAT;
  const float* zi = z_imag   + (size_t)m * DLAT;
  v2f* outc = (v2f*)out_z + (size_t)m * DLAT;

  float s = 0.f;
  for (int j = lane; j < DLAT; j += 32) {
    float cr = c[j], ci = c[j + DLAT];
    float dr = cr - zr[j], di = ci - zi[j];
    s += dr * dr + di * di;
    v2f o; o.x = cr; o.y = ci;
    outc[j] = o;
  }
  #pragma unroll
  for (int off = 16; off >= 1; off >>= 1) s += __shfl_xor(s, off, 32);
  if (lane == 0) out_loss[m] = 1.01f * s * (1.0f / (float)KD);
}

// ---------------------------------------------------------------------------
extern "C" void kernel_launch(void* const* d_in, const int* in_sizes, int n_in,
                              void* d_out, int out_size, void* d_ws, size_t ws_size,
                              hipStream_t stream) {
  (void)in_sizes; (void)n_in; (void)out_size; (void)ws_size;
  const float* z_real    = (const float*)d_in[0];
  const float* z_imag    = (const float*)d_in[1];
  const int*   prev      = (const int*)  d_in[2];
  const float* codebook  = (const float*)d_in[3];
  const float* adjacency = (const float*)d_in[4];

  float* out = (float*)d_out;
  // output tuple layout: z_complex (2*M*D floats) | loss (M) | min_indices (M, int32 bits)
  const size_t ZOUT = (size_t)2 * M_TOTAL * DLAT;
  float* out_loss = out + ZOUT;
  int*   idx_out  = (int*)(out + ZOUT + M_TOTAL);

  // workspace: z_hi | z_lo | c_hi | c_lo | cnorm  (~72 MB)
  __bf16* z_hi = (__bf16*)d_ws;
  __bf16* z_lo = z_hi + (size_t)M_TOTAL * KD;
  __bf16* c_hi = z_lo + (size_t)M_TOTAL * KD;
  __bf16* c_lo = c_hi + (size_t)NSYM * KD;
  float*  cnorm = (float*)(c_lo + (size_t)NSYM * KD);

  convert_z_kernel<<<(M_TOTAL * KD / 4) / 256, 256, 0, stream>>>(z_real, z_imag, z_hi, z_lo);
  convert_c_kernel<<<(NSYM * KD / 4) / 256, 256, 0, stream>>>(codebook, c_hi, c_lo);
  cnorm_kernel<<<NSYM / 256, 256, 0, stream>>>(codebook, cnorm);

  vq_argmin_kernel<<<M_TOTAL / 128, 256, 2 * BUFBYTES, stream>>>(
      z_hi, z_lo, c_hi, c_lo, prev, adjacency, cnorm, idx_out);

  finalize_kernel<<<M_TOTAL / 8, 256, 0, stream>>>(
      z_real, z_imag, codebook, idx_out, out, out_loss);
}
